// HierarchicalRefinementDecoder_38448547234595
// MI455X (gfx1250) — matrix operand order repaired and checked by reading, not compiled, not run on hardware
//
#include <hip/hip_runtime.h>
#include <hip/hip_bf16.h>

typedef __attribute__((ext_vector_type(16))) _Float16 v16h;
typedef __attribute__((ext_vector_type(8)))  float    v8f;

union AFrag { v16h vec; unsigned int u[8]; };

__device__ __forceinline__ v8f wmma_f16(const AFrag& a, const AFrag& b, v8f c) {
  return __builtin_amdgcn_wmma_f32_16x16x32_f16(false, a.vec, false, b.vec, (short)0, c, false, false);
}

// gfx1250 async global->LDS copy (16B per lane), tracked by ASYNCcnt.
__device__ __forceinline__ void async_copy_b128(void* lds, const void* gaddr) {
  unsigned ldsOff = (unsigned)(uintptr_t)lds;                   // addr[31:0] = LDS offset
  unsigned long long ga = (unsigned long long)(uintptr_t)gaddr;
  asm volatile("global_load_async_to_lds_b128 %0, %1, off"
               :: "v"(ldsOff), "v"(ga) : "memory");
}
__device__ __forceinline__ void async_wait0() {
#if __has_builtin(__builtin_amdgcn_s_wait_asynccnt)
  __builtin_amdgcn_s_wait_asynccnt(0);
#else
  asm volatile("s_wait_asynccnt 0" ::: "memory");
#endif
}

#define GSL(i, total) for (long long i = (long long)blockIdx.x * blockDim.x + threadIdx.x; \
                           i < (total); i += (long long)gridDim.x * blockDim.x)

// ---------------------------------------------------------------------------
// Generic WMMA GEMM: C[M,N] = act(A_f16[M,K] @ W_f16[K,N] + bias)
// 64x64 tile per 128-thread block, 4 waves in 2x2 arrangement.
// Requires: M % 64 == 0, K % 32 == 0 (all call sites satisfy this).
// A tile is staged via async global->LDS; W tile via transposing LDS stores.
// ---------------------------------------------------------------------------
#define GLDA 40

__global__ __launch_bounds__(128)
void gemm_wmma_kernel(const _Float16* __restrict__ A, const _Float16* __restrict__ W,
                      const float* __restrict__ bias, int nbias,
                      float* __restrict__ Cf, _Float16* __restrict__ Ch,
                      int M, int N, int K, int act)
{
  extern __shared__ char smem_g[];
  _Float16* sA = (_Float16*)smem_g;      // [64][GLDA] row-major (m,k)
  _Float16* sB = sA + 64 * GLDA;         // [64][GLDA] transposed (n,k)
  const int tid = threadIdx.x;
  const int wave = tid >> 5, lane = tid & 31;
  const int h = lane >> 4, ln = lane & 15;
  const int wr = wave >> 1, wc = wave & 1;
  const int mBase = blockIdx.y * 64, nBase = blockIdx.x * 64;

  v8f acc[2][2] = {};

  for (int k0 = 0; k0 < K; k0 += 32) {
    __syncthreads();
#pragma unroll
    for (int it = 0; it < 2; ++it) {                 // A tile 64x32: async -> LDS
      int e = (tid + it * 128) * 8;
      int r = e >> 5, c = e & 31;
      const _Float16* src = A + (size_t)(mBase + r) * K + k0 + c;
      async_copy_b128(sA + r * GLDA + c, src);
      if (k0 + 32 < K) __builtin_prefetch(src + 32, 0, 1);
    }
#pragma unroll
    for (int it = 0; it < 2; ++it) {                 // W tile 32x64, store transposed
      int e = (tid + it * 128) * 8;
      int r = e >> 6, c = e & 63;
      _Float16 tmp[8];
      const _Float16* src = W + (size_t)(k0 + r) * N + nBase + c;
      if (nBase + c + 8 <= N) {
        *(uint4*)tmp = *(const uint4*)src;
        if (k0 + 32 < K) __builtin_prefetch(src + (size_t)32 * N, 0, 1);
      } else {
#pragma unroll
        for (int j = 0; j < 8; ++j) tmp[j] = (nBase + c + j < N) ? src[j] : (_Float16)0.f;
      }
#pragma unroll
      for (int j = 0; j < 8; ++j) sB[(c + j) * GLDA + r] = tmp[j];
    }
    async_wait0();
    __syncthreads();

    AFrag af[2], bf[2];
#pragma unroll
    for (int mi = 0; mi < 2; ++mi) {
      int m = wr * 32 + mi * 16 + ln;
#pragma unroll
      for (int vv = 0; vv < 8; ++vv) {
        int kk = ((vv & 4) << 2) + ((vv & 3) << 1) + (h << 3);   // A-frag K layout
        af[mi].u[vv] = *(const unsigned int*)(sA + m * GLDA + kk);
      }
    }
#pragma unroll
    for (int ni = 0; ni < 2; ++ni) {
      int n = wc * 32 + ni * 16 + ln;
#pragma unroll
      for (int vv = 0; vv < 8; ++vv) {
        int kk = ((vv & 4) << 2) + ((vv & 3) << 1) + (h << 3);   // B-frag K layout
        bf[ni].u[vv] = *(const unsigned int*)(sB + n * GLDA + kk);
      }
    }
#pragma unroll
    for (int mi = 0; mi < 2; ++mi)
#pragma unroll
      for (int ni = 0; ni < 2; ++ni)
        acc[mi][ni] = wmma_f16(af[mi], bf[ni], acc[mi][ni]);
  }

#pragma unroll
  for (int mi = 0; mi < 2; ++mi)
#pragma unroll
    for (int ni = 0; ni < 2; ++ni) {
      int col = nBase + wc * 32 + ni * 16 + ln;
      float bv = (bias != nullptr && col < nbias) ? bias[col] : 0.f;
#pragma unroll
      for (int i = 0; i < 8; ++i) {
        int row = mBase + wr * 32 + mi * 16 + i + h * 8;
        if (col < N) {
          float v = acc[mi][ni][i] + bv;
          if (act) v = fmaxf(v, 0.f);
          if (Cf) Cf[(size_t)row * N + col] = v;
          if (Ch) Ch[(size_t)row * N + col] = (_Float16)v;
        }
      }
    }
}

// ---------------------------------------------------------------------------
// Fused cross-attention with distance bias + online softmax (flash style).
// 16 query rows per 128-thread block; 64 keys per iteration (16 per wave).
// out = softmax(Q K^T / 16 - 0.5*dist) @ V + new_feat   (f16 out)
// ---------------------------------------------------------------------------
#define ALDV 72
#define ALDP 72

__global__ __launch_bounds__(128)
void attn_kernel(const _Float16* __restrict__ Q, const _Float16* __restrict__ Kg,
                 const _Float16* __restrict__ Vt, const float* __restrict__ dxyz,
                 const float* __restrict__ sxyz, const _Float16* __restrict__ nf,
                 _Float16* __restrict__ out, int N2, int Msp)
{
  extern __shared__ char smem_a[];
  _Float16* sVt = (_Float16*)smem_a;          // [256][ALDV] (feat, key)
  _Float16* sP  = sVt + 256 * ALDV;           // [16][ALDP]  (qrow, key)
  float* sXd   = (float*)(sP + 16 * ALDP);    // [16][4]
  float* sM    = sXd + 64;                    // [16] running max
  float* sL    = sM + 16;                     // [16] running sum
  float* sCorr = sL + 16;                     // [16]
  float* sRedM = sCorr + 16;                  // [4][16]
  float* sRedS = sRedM + 64;                  // [4][16]

  const int tid = threadIdx.x;
  const int wave = tid >> 5, lane = tid & 31;
  const int h = lane >> 4, ln = lane & 15;
  const int b = blockIdx.y;
  const int qBase = blockIdx.x * 16;

  // Q fragments for the whole 256-wide row, kept in registers for all key blocks
  AFrag aq[8];
  {
    const _Float16* qrow = Q + ((size_t)b * N2 + qBase + ln) * 256;
#pragma unroll
    for (int kc = 0; kc < 8; ++kc)
#pragma unroll
      for (int vv = 0; vv < 8; ++vv) {
        int kk = kc * 32 + ((vv & 4) << 2) + ((vv & 3) << 1) + (h << 3);
        aq[kc].u[vv] = *(const unsigned int*)(qrow + kk);
      }
  }
  if (tid < 16) { sM[tid] = -3.0e38f; sL[tid] = 0.f; }
  if (tid < 48) sXd[(tid / 3) * 4 + (tid % 3)] = dxyz[((size_t)b * N2 + qBase) * 3 + tid];

  v8f oacc[4] = {};

  for (int kb = 0; kb < Msp; kb += 64) {
    __syncthreads();
    // stage V^T chunk [256 feat][64 keys] via async global->LDS
    for (int i = tid * 8; i < 256 * 64; i += 128 * 8) {
      int f = i >> 6, c = i & 63;
      const _Float16* src = Vt + ((size_t)b * 256 + f) * Msp + kb + c;
      async_copy_b128(sVt + f * ALDV + c, src);
      if (kb + 64 < Msp) __builtin_prefetch(src + 64, 0, 1);
    }
    // scores for this wave's 16 keys, K rows straight from global (L2 resident)
    const _Float16* krow = Kg + ((size_t)b * Msp + kb + wave * 16 + ln) * 256;
    if (kb + 64 < Msp) __builtin_prefetch(krow + 64 * 256, 0, 1);
    v8f s = {};
#pragma unroll
    for (int kc = 0; kc < 8; ++kc) {
      AFrag bk;
#pragma unroll
      for (int vv = 0; vv < 8; ++vv) {
        int kk = kc * 32 + ((vv & 4) << 2) + ((vv & 3) << 1) + (h << 3);
        bk.u[vv] = *(const unsigned int*)(krow + kk);
      }
      s = wmma_f16(aq[kc], bk, s);
    }
    int key = kb + wave * 16 + ln;
    float kx = sxyz[((size_t)b * Msp + key) * 3 + 0];
    float ky = sxyz[((size_t)b * Msp + key) * 3 + 1];
    float kz = sxyz[((size_t)b * Msp + key) * 3 + 2];
    float sv[8];
#pragma unroll
    for (int i = 0; i < 8; ++i) {
      int row = i + 8 * h;
      float dx = sXd[row * 4 + 0] - kx, dy = sXd[row * 4 + 1] - ky, dz = sXd[row * 4 + 2] - kz;
      float d2 = dx * dx + dy * dy + dz * dz;
      sv[i] = s[i] * 0.0625f - 0.5f * sqrtf(fmaxf(d2, 1e-12f));
    }
    // per-row max over this wave's 16 keys (cols live across 16 lanes)
#pragma unroll
    for (int i = 0; i < 8; ++i) {
      float v = sv[i];
      v = fmaxf(v, __shfl_xor(v, 1, 16)); v = fmaxf(v, __shfl_xor(v, 2, 16));
      v = fmaxf(v, __shfl_xor(v, 4, 16)); v = fmaxf(v, __shfl_xor(v, 8, 16));
      if (ln == 0) sRedM[wave * 16 + i + 8 * h] = v;
    }
    __syncthreads();
    if (tid < 16) {
      float tm = fmaxf(fmaxf(sRedM[tid], sRedM[16 + tid]), fmaxf(sRedM[32 + tid], sRedM[48 + tid]));
      float mo = sM[tid], mn = fmaxf(mo, tm);
      sCorr[tid] = __expf(mo - mn);
      sM[tid] = mn;
    }
    __syncthreads();
#pragma unroll
    for (int i = 0; i < 8; ++i) {
      int row = i + 8 * h;
      float p = __expf(sv[i] - sM[row]);
      sP[row * ALDP + wave * 16 + ln] = (_Float16)p;
      float v = p;
      v += __shfl_xor(v, 1, 16); v += __shfl_xor(v, 2, 16);
      v += __shfl_xor(v, 4, 16); v += __shfl_xor(v, 8, 16);
      if (ln == 0) sRedS[wave * 16 + row] = v;
    }
#pragma unroll
    for (int t = 0; t < 4; ++t)
#pragma unroll
      for (int i = 0; i < 8; ++i) oacc[t][i] *= sCorr[i + 8 * h];
    async_wait0();
    __syncthreads();
    if (tid < 16)
      sL[tid] = sL[tid] * sCorr[tid] + sRedS[tid] + sRedS[16 + tid] + sRedS[32 + tid] + sRedS[48 + tid];
    // AV: P[16x64] @ V[64 x 64cols-per-wave]
#pragma unroll
    for (int kc = 0; kc < 2; ++kc) {
      AFrag ap;
#pragma unroll
      for (int vv = 0; vv < 8; ++vv) {
        int kk = kc * 32 + ((vv & 4) << 2) + ((vv & 3) << 1) + (h << 3);
        ap.u[vv] = *(const unsigned int*)(sP + ln * ALDP + kk);
      }
#pragma unroll
      for (int t = 0; t < 4; ++t) {
        AFrag bv;
#pragma unroll
        for (int vv = 0; vv < 8; ++vv) {
          int kk = kc * 32 + ((vv & 4) << 2) + ((vv & 3) << 1) + (h << 3);
          bv.u[vv] = *(const unsigned int*)(sVt + (wave * 64 + t * 16 + ln) * ALDV + kk);
        }
        oacc[t] = wmma_f16(ap, bv, oacc[t]);
      }
    }
  }
  __syncthreads();
#pragma unroll
  for (int t = 0; t < 4; ++t)
#pragma unroll
    for (int i = 0; i < 8; ++i) {
      int row = i + 8 * h, col = wave * 64 + t * 16 + ln;
      size_t gi = ((size_t)b * N2 + qBase + row) * 256 + col;
      float v = oacc[t][i] / sL[row] + (float)nf[gi];
      out[gi] = (_Float16)v;
    }
}

// ---------------------------------------------------------------------------
// Elementwise glue kernels
// ---------------------------------------------------------------------------
__global__ void convert_weight_kernel(const float* __restrict__ w, _Float16* __restrict__ o,
                                      int Ko, int No, int Kp, int Np) {
  long long total = (long long)Kp * Np;
  GSL(i, total) {
    int k = (int)(i / Np), n = (int)(i % Np);
    o[i] = (k < Ko && n < No) ? (_Float16)w[(long long)k * No + n] : (_Float16)0.f;
  }
}
__global__ void convert_f16_kernel(const float* __restrict__ x, _Float16* __restrict__ y, long long n) {
  GSL(i, n) y[i] = (_Float16)x[i];
}
__global__ void fold1_pack_kernel(const float* __restrict__ gf, _Float16* __restrict__ o) {
  long long total = 8192LL * 288;
  GSL(i, total) {
    long long row = i / 288; int c = (int)(i % 288);
    int b = (int)(row >> 10), p = (int)(row & 1023);
    float v;
    if (c < 256) v = gf[(b << 8) + c];
    else if (c == 256) v = -1.f + 2.f * (float)(p >> 5) / 31.f;
    else if (c == 257) v = -1.f + 2.f * (float)(p & 31) / 31.f;
    else v = 0.f;
    o[i] = (_Float16)v;
  }
}
__global__ void fold2_pack_kernel(const float* __restrict__ gf, const float* __restrict__ f1,
                                  _Float16* __restrict__ o) {
  long long total = 8192LL * 288;
  GSL(i, total) {
    long long row = i / 288; int c = (int)(i % 288);
    int b = (int)(row >> 10);
    float v;
    if (c < 256) v = gf[(b << 8) + c];
    else if (c < 259) v = f1[row * 16 + (c - 256)];
    else v = 0.f;
    o[i] = (_Float16)v;
  }
}
__global__ void bcast_feat_kernel(const float* __restrict__ gf, _Float16* __restrict__ o) {
  long long total = 8192LL * 256;
  GSL(i, total) {
    long long row = i >> 8; int c = (int)(i & 255);
    int b = (int)(row >> 10);
    o[i] = (_Float16)gf[(b << 8) + c];
  }
}
__global__ void compact_xyz_kernel(const float* __restrict__ s, float* __restrict__ d, long long rows) {
  GSL(i, rows * 3) { long long r = i / 3; int c = (int)(i % 3); d[i] = s[r * 16 + c]; }
}
__global__ void pack_combined_kernel(const _Float16* __restrict__ feat, const float* __restrict__ gf,
                                     _Float16* __restrict__ o, int rowsPerB, long long total) {
  GSL(i, total) {
    long long row = i >> 9; int c = (int)(i & 511);
    int b = (int)(row / rowsPerB);
    o[i] = (c < 256) ? feat[(row << 8) + c] : (_Float16)gf[(b << 8) + (c - 256)];
  }
}
__global__ void pack_qin_kernel(const float* __restrict__ xyz, const _Float16* __restrict__ feat,
                                _Float16* __restrict__ o, long long total) {
  GSL(i, total) {
    long long row = i / 288; int c = (int)(i % 288);
    if (c < 3) o[i] = (_Float16)xyz[row * 3 + c];
    else if (c < 259) o[i] = feat[(row << 8) + (c - 3)];
    else o[i] = (_Float16)0.f;
  }
}
__global__ void pack_kvin_kernel(const float* __restrict__ xyz, const float* __restrict__ feat,
                                 _Float16* __restrict__ o, long long total) {
  GSL(i, total) {
    long long row = i / 288; int c = (int)(i % 288);
    float v;
    if (c < 3) v = xyz[row * 3 + c];
    else if (c < 259) v = feat[(row << 8) + (c - 3)];
    else v = 0.f;
    o[i] = (_Float16)v;
  }
}
__global__ void newxyz_kernel(const float* __restrict__ xyz, const float* __restrict__ offs,
                              float* __restrict__ nxyz, long long R) {
  GSL(i, R * 6) {
    long long r = i / 6; int rem = (int)(i % 6);
    int s2 = rem / 3, d = rem % 3;
    nxyz[(r * 2 + s2) * 3 + d] = xyz[r * 3 + d] + 0.1f * offs[r * 16 + s2 * 3 + d];
  }
}
__global__ void transpose_v_kernel(const _Float16* __restrict__ V, _Float16* __restrict__ Vt,
                                   long long total, int Msp) {
  GSL(i, total) {
    long long b = i / ((long long)Msp * 256);
    long long rem = i - b * (long long)Msp * 256;
    int k = (int)(rem >> 8), f = (int)(rem & 255);
    Vt[(b * 256 + f) * (long long)Msp + k] = V[i];
  }
}
__global__ void copy_out_kernel(const float* __restrict__ x, float* __restrict__ o, long long n) {
  GSL(i, n) o[i] = x[i];
}

// ---------------------------------------------------------------------------
// Host orchestration
// ---------------------------------------------------------------------------
extern "C" void kernel_launch(void* const* d_in, const int* in_sizes, int n_in,
                              void* d_out, int out_size, void* d_ws, size_t ws_size,
                              hipStream_t stream) {
  (void)in_sizes; (void)n_in; (void)out_size; (void)ws_size;
  const int Bn = 8, MS = 2048;
  const float* partial_xyz  = (const float*)d_in[0];
  const float* partial_feat = (const float*)d_in[1];
  const float* global_feat  = (const float*)d_in[2];

  int pi = 3;
  auto nxt = [&]() { return (const float*)d_in[pi++]; };
  const float *f1b0=nxt(),*f1w0=nxt(),*f1b1=nxt(),*f1w1=nxt(),*f1b2=nxt(),*f1w2=nxt();
  const float *f2b0=nxt(),*f2w0=nxt(),*f2b1=nxt(),*f2w1=nxt(),*f2b2=nxt(),*f2w2=nxt();
  const float *kb_[4],*kw_[4],*ob_[4],*ow_[4],*qb_[4],*qw_[4],*vb_[4],*vw_[4];
  for (int s = 0; s < 4; ++s) { kb_[s]=nxt();kw_[s]=nxt();ob_[s]=nxt();ow_[s]=nxt();
                                qb_[s]=nxt();qw_[s]=nxt();vb_[s]=nxt();vw_[s]=nxt(); }
  const float *ftb0[4],*ftw0[4],*ftb1[4],*ftw1[4];
  const float *ofb0[4],*ofw0[4],*ofb1[4],*ofw1[4],*ofb2[4],*ofw2[4];
  for (int s = 0; s < 4; ++s) { ftb0[s]=nxt();ftw0[s]=nxt();ftb1[s]=nxt();ftw1[s]=nxt();
                                ofb0[s]=nxt();ofw0[s]=nxt();ofb1[s]=nxt();ofw1[s]=nxt();
                                ofb2[s]=nxt();ofw2[s]=nxt(); }

  // workspace bump allocator
  char* ws = (char*)d_ws; size_t off = 0;
  auto alloc = [&](size_t bytes) { void* p = ws + off; off += (bytes + 255) & ~(size_t)255; return p; };
  const size_t maxR2 = (size_t)Bn * 16384, maxR = (size_t)Bn * 8192;
  _Float16* curr_feat = (_Float16*)alloc(maxR2 * 256 * 2);
  float* xyzA = (float*)alloc(maxR2 * 3 * 4);
  float* xyzB = (float*)alloc(maxR2 * 3 * 4);
  _Float16* bigA = (_Float16*)alloc(maxR2 * 288 * 2);   // combined / qpack / attn-in
  _Float16* bigB = (_Float16*)alloc(maxR  * 512 * 2);   // new_feat
  _Float16* bigC = (_Float16*)alloc(maxR2 * 256 * 2);   // h2 / Q
  _Float16* h1   = (_Float16*)alloc(maxR  * 256 * 2);
  float* offs    = (float*)alloc(maxR * 16 * 4);
  _Float16* kvpack = (_Float16*)alloc((size_t)Bn * MS * 288 * 2);
  _Float16* K16  = (_Float16*)alloc((size_t)Bn * MS * 256 * 2);
  _Float16* V16  = (_Float16*)alloc((size_t)Bn * MS * 256 * 2);
  _Float16* Vt16 = (_Float16*)alloc((size_t)Bn * 256 * MS * 2);
  _Float16* pf16 = (_Float16*)alloc((size_t)Bn * MS * 256 * 2);

  auto ewg = [&](long long total) {
    long long b = (total + 255) / 256; if (b > 1048576) b = 1048576;
    return dim3((unsigned)b);
  };
  auto cw = [&](const float* src, int Ko, int No, int Kp, int Np) {
    _Float16* dst = (_Float16*)alloc((size_t)Kp * Np * 2);
    long long tot = (long long)Kp * Np;
    convert_weight_kernel<<<ewg(tot), 256, 0, stream>>>(src, dst, Ko, No, Kp, Np);
    return dst;
  };
  const size_t gemmSm = (size_t)2 * 64 * GLDA * sizeof(_Float16);
  auto gemm = [&](const _Float16* A, const _Float16* W, const float* bias, int nbias,
                  float* Cf, _Float16* Ch, int M, int N, int K, int act) {
    dim3 g((N + 63) / 64, (M + 63) / 64);
    gemm_wmma_kernel<<<g, 128, gemmSm, stream>>>(A, W, bias, nbias, Cf, Ch, M, N, K, act);
  };

  // ---- weight conversion (padded f16) ----
  _Float16 *wf1_0 = cw(f1w0, 258, 256, 288, 256), *wf1_1 = cw(f1w1, 256, 256, 256, 256),
           *wf1_2 = cw(f1w2, 256, 3, 256, 16);
  _Float16 *wf2_0 = cw(f2w0, 259, 256, 288, 256), *wf2_1 = cw(f2w1, 256, 256, 256, 256),
           *wf2_2 = cw(f2w2, 256, 3, 256, 16);
  _Float16 *wof0[4], *wof1[4], *wof2[4], *wft0[4], *wft1[4], *wq[4], *wk[4], *wv[4], *wo[4];
  for (int s = 0; s < 4; ++s) {
    wof0[s] = cw(ofw0[s], 512, 256, 512, 256);
    wof1[s] = cw(ofw1[s], 256, 256, 256, 256);
    wof2[s] = cw(ofw2[s], 256, 6, 256, 16);
    wft0[s] = cw(ftw0[s], 512, 256, 512, 256);
    wft1[s] = cw(ftw1[s], 256, 512, 256, 512);
    wq[s] = cw(qw_[s], 259, 256, 288, 256);
    wk[s] = cw(kw_[s], 259, 256, 288, 256);
    wv[s] = cw(vw_[s], 256, 256, 256, 256);
    wo[s] = cw(ow_[s], 256, 256, 256, 256);
  }
  convert_f16_kernel<<<ewg((long long)Bn*MS*256), 256, 0, stream>>>(partial_feat, pf16, (long long)Bn*MS*256);
  pack_kvin_kernel<<<ewg((long long)Bn*MS*288), 256, 0, stream>>>(partial_xyz, partial_feat, kvpack,
                                                                  (long long)Bn*MS*288);

  // ---- folding decoder ----
  const int R0 = Bn * 1024;
  fold1_pack_kernel<<<ewg((long long)R0*288), 256, 0, stream>>>(global_feat, bigA);
  gemm(bigA, wf1_0, f1b0, 256, nullptr, h1,   R0, 256, 288, 1);
  gemm(h1,   wf1_1, f1b1, 256, nullptr, bigC, R0, 256, 256, 1);
  gemm(bigC, wf1_2, f1b2, 3,   offs,   nullptr, R0, 16, 256, 0);
  fold2_pack_kernel<<<ewg((long long)R0*288), 256, 0, stream>>>(global_feat, offs, bigA);
  gemm(bigA, wf2_0, f2b0, 256, nullptr, h1,   R0, 256, 288, 1);
  gemm(h1,   wf2_1, f2b1, 256, nullptr, bigC, R0, 256, 256, 1);
  gemm(bigC, wf2_2, f2b2, 3,   offs,   nullptr, R0, 16, 256, 0);
  compact_xyz_kernel<<<ewg((long long)R0*3), 256, 0, stream>>>(offs, xyzA, R0);
  bcast_feat_kernel<<<ewg((long long)R0*256), 256, 0, stream>>>(global_feat, curr_feat);

  // ---- refinement stages ----
  const size_t attnSm = (size_t)(256 * ALDV + 16 * ALDP) * sizeof(_Float16) + 240 * sizeof(float);
  int N = 1024;
  for (int s = 0; s < 4; ++s) {
    const long long R = (long long)Bn * N, R2 = 2 * R;
    pack_combined_kernel<<<ewg(R*512), 256, 0, stream>>>(curr_feat, global_feat, bigA, N, R*512);
    gemm(bigA, wof0[s], ofb0[s], 256, nullptr, h1,   (int)R, 256, 512, 1);
    gemm(h1,   wof1[s], ofb1[s], 256, nullptr, bigC, (int)R, 256, 256, 1);
    gemm(bigC, wof2[s], ofb2[s], 6,   offs,   nullptr, (int)R, 16, 256, 0);
    gemm(bigA, wft0[s], ftb0[s], 256, nullptr, h1,   (int)R, 256, 512, 1);
    gemm(h1,   wft1[s], ftb1[s], 512, nullptr, bigB, (int)R, 512, 256, 0);
    newxyz_kernel<<<ewg(R*6), 256, 0, stream>>>(xyzA, offs, xyzB, R);
    pack_qin_kernel<<<ewg(R2*288), 256, 0, stream>>>(xyzB, bigB, bigA, R2*288);
    gemm(bigA, wq[s], qb_[s], 256, nullptr, bigC, (int)R2, 256, 288, 0);      // Q
    gemm(kvpack, wk[s], kb_[s], 256, nullptr, K16, Bn*MS, 256, 288, 0);       // K
    gemm(pf16,   wv[s], vb_[s], 256, nullptr, V16, Bn*MS, 256, 256, 0);       // V
    transpose_v_kernel<<<ewg((long long)Bn*MS*256), 256, 0, stream>>>(V16, Vt16, (long long)Bn*MS*256, MS);
    attn_kernel<<<dim3((2*N)/16, Bn), 128, attnSm, stream>>>(bigC, K16, Vt16, xyzB, partial_xyz,
                                                             bigB, bigA, 2*N, MS);
    gemm(bigA, wo[s], ob_[s], 256, nullptr, curr_feat, (int)R2, 256, 256, 0);
    float* t = xyzA; xyzA = xyzB; xyzB = t;
    N *= 2;
  }
  copy_out_kernel<<<ewg((long long)Bn*16384*3), 256, 0, stream>>>(xyzA, (float*)d_out,
                                                                  (long long)Bn*16384*3);
}
